// HSTU_86955907875122
// MI455X (gfx1250) — matrix operand order repaired and checked by reading, not compiled, not run on hardware
//
#include <hip/hip_runtime.h>
#include <math.h>

#define DD 512
#define HH 8
#define DHD 64
#define FFD 2048
#define VVD 32000
#define SSD 1024
#define BBD 4
#define LLD 6
#define MMD (BBD * SSD)   // 4096 rows

typedef __bf16 bf16;
typedef __attribute__((ext_vector_type(16))) __bf16          v16bf;
typedef __attribute__((ext_vector_type(8)))  float           v8f;
typedef __attribute__((ext_vector_type(8)))  unsigned short  ushort8;

union Frag {
    v16bf   v;
    bf16    e[16];
    ushort8 h[2];    // two 128-bit halves -> global_load_b128 / ds_load_b128
};

__device__ __forceinline__ v8f zero8() {
    v8f z;
    #pragma unroll
    for (int i = 0; i < 8; ++i) z[i] = 0.0f;
    return z;
}

// ---------------------------------------------------------------------------
// Weight transpose + convert: Wt[n,k] (bf16) = W[k,n] (f32)
// 32x32 tiles through LDS so both loads (along N) and stores (along K)
// are coalesced. One-shot per weight per launch; pure bandwidth.
// ---------------------------------------------------------------------------
__global__ __launch_bounds__(256) void convt_kernel(const float* __restrict__ W,
                                                    bf16* __restrict__ Wt,
                                                    int K, int N) {
    __shared__ float tile[32][33];
    const int tx = threadIdx.x & 31;
    const int ty = threadIdx.x >> 5;          // 0..7
    const int nb = blockIdx.x * 32;
    const int kb = blockIdx.y * 32;
    #pragma unroll
    for (int r = 0; r < 32; r += 8)
        tile[ty + r][tx] = W[(size_t)(kb + ty + r) * N + nb + tx];
    __syncthreads();
    #pragma unroll
    for (int r = 0; r < 32; r += 8)
        Wt[(size_t)(nb + ty + r) * K + kb + tx] = (bf16)tile[tx][ty + r];
}

// ---------------------------------------------------------------------------
// Embedding: h[b,s,:] = emb[x[b,s]] * sqrt(D) + pos[s]  (f32 + bf16 copies)
// ---------------------------------------------------------------------------
__global__ __launch_bounds__(256) void embed_kernel(const int* __restrict__ x,
                                                    const float* __restrict__ emb,
                                                    const float* __restrict__ pos,
                                                    float* __restrict__ h32,
                                                    bf16* __restrict__ h16) {
    int bs = blockIdx.x;
    int tok = x[bs];
    int sp = bs % SSD;
    const float sq = 22.62741699796952f; // sqrt(512)
    for (int d = threadIdx.x; d < DD; d += blockDim.x) {
        float v = emb[(size_t)tok * DD + d] * sq + pos[(size_t)sp * DD + d];
        h32[(size_t)bs * DD + d] = v;
        h16[(size_t)bs * DD + d] = (bf16)v;
    }
}

// ---------------------------------------------------------------------------
// Residual + LayerNorm: out = LN(x + resid) * w + b
// resid / out32 / out16 individually optional (nullptr).
// ---------------------------------------------------------------------------
__global__ __launch_bounds__(256) void ln_kernel(const float* __restrict__ x,
                                                 const float* __restrict__ resid,
                                                 const float* __restrict__ w,
                                                 const float* __restrict__ b,
                                                 float* __restrict__ out32,
                                                 bf16* __restrict__ out16) {
    __shared__ float red[256];
    int row = blockIdx.x;
    int t = threadIdx.x;
    size_t base = (size_t)row * DD;
    float v0 = x[base + t]       + (resid ? resid[base + t]       : 0.0f);
    float v1 = x[base + t + 256] + (resid ? resid[base + t + 256] : 0.0f);
    red[t] = v0 + v1;
    __syncthreads();
    #pragma unroll
    for (int o = 128; o > 0; o >>= 1) {
        if (t < o) red[t] += red[t + o];
        __syncthreads();
    }
    float mu = red[0] * (1.0f / DD);
    __syncthreads();
    float d0 = v0 - mu, d1 = v1 - mu;
    red[t] = d0 * d0 + d1 * d1;
    __syncthreads();
    #pragma unroll
    for (int o = 128; o > 0; o >>= 1) {
        if (t < o) red[t] += red[t + o];
        __syncthreads();
    }
    float rstd = rsqrtf(red[0] * (1.0f / DD) + 1e-5f);
    float r0 = d0 * rstd * w[t]       + b[t];
    float r1 = d1 * rstd * w[t + 256] + b[t + 256];
    if (out32) { out32[base + t] = r0; out32[base + t + 256] = r1; }
    if (out16) { out16[base + t] = (bf16)r0; out16[base + t + 256] = (bf16)r1; }
}

// ---------------------------------------------------------------------------
// Pure-bf16 WMMA GEMM:  C[M,N] = act(A[M,K] @ Wt[N,K]^T + bias)
// A: bf16 row-major (M,K). Wt: bf16 transposed weights (N,K) so every
// fragment is two contiguous b128 loads per lane. Wave tile = 32x64:
// 2 A-frags, 4 shared B-frags, 8 v_wmma_f32_16x16x32_bf16 per K-step.
// ---------------------------------------------------------------------------
template <bool DO_GELU, bool OUT_BF16>
__global__ __launch_bounds__(256) void gemm_wmma(const bf16* __restrict__ A,
                                                 const bf16* __restrict__ Wt,
                                                 const float* __restrict__ bias,
                                                 void* __restrict__ C_,
                                                 int M, int N, int K) {
    const int lane = threadIdx.x & 31;
    const int wid  = blockIdx.x * (blockDim.x >> 5) + (threadIdx.x >> 5);
    const int ntiles = N >> 6;   // 64-wide
    const int mtiles = M >> 5;   // 32-high
    if (wid >= mtiles * ntiles) return;   // wave-uniform exit (EXEC stays full)
    const int mt = wid / ntiles;
    const int nt = wid % ntiles;

    const int colq = lane & 15;
    const int grpA = (lane < 16) ? 0 : 8;    // A-fragment K offset per lane group
    const int grpB = (lane < 16) ? 0 : 8;    // B-fragment K offset per lane group
    const int grpC = (lane < 16) ? 0 : 8;    // C/D row offset per lane group

    const bf16* arow0 = A + (size_t)(mt * 32 + colq) * K + grpA;
    const bf16* arow1 = A + (size_t)(mt * 32 + 16 + colq) * K + grpA;
    const bf16* wbase = Wt + (size_t)(nt * 64 + colq) * K + grpB;

    v8f acc[2][4];
    #pragma unroll
    for (int g = 0; g < 2; ++g)
        #pragma unroll
        for (int f = 0; f < 4; ++f) acc[g][f] = zero8();

    for (int k0 = 0; k0 < K; k0 += 32) {
        Frag a0, a1;
        a0.h[0] = *(const ushort8*)(arow0 + k0);
        a0.h[1] = *(const ushort8*)(arow0 + k0 + 16);
        a1.h[0] = *(const ushort8*)(arow1 + k0);
        a1.h[1] = *(const ushort8*)(arow1 + k0 + 16);
        if (k0 + 32 < K) {
            __builtin_prefetch(arow0 + k0 + 32, 0, 3);
            __builtin_prefetch(arow1 + k0 + 32, 0, 3);
        }
        #pragma unroll
        for (int f = 0; f < 4; ++f) {
            Frag bfr;
            const bf16* wp = wbase + (size_t)(f * 16) * K + k0;
            bfr.h[0] = *(const ushort8*)(wp);
            bfr.h[1] = *(const ushort8*)(wp + 16);
            acc[0][f] = __builtin_amdgcn_wmma_f32_16x16x32_bf16(
                false, a0.v, false, bfr.v, (short)0, acc[0][f], false, false);
            acc[1][f] = __builtin_amdgcn_wmma_f32_16x16x32_bf16(
                false, a1.v, false, bfr.v, (short)0, acc[1][f], false, false);
        }
    }

    #pragma unroll
    for (int g = 0; g < 2; ++g)
        #pragma unroll
        for (int f = 0; f < 4; ++f) {
            int col = nt * 64 + f * 16 + colq;
            float bv = bias ? bias[col] : 0.0f;
            #pragma unroll
            for (int r = 0; r < 8; ++r) {
                int orow = mt * 32 + g * 16 + r + grpC;
                float v = acc[g][f][r] + bv;
                if (DO_GELU) v = 0.5f * v * (1.0f + erff(v * 0.70710678118654752f));
                if (OUT_BF16) ((bf16*)C_)[(size_t)orow * N + col] = (bf16)v;
                else          ((float*)C_)[(size_t)orow * N + col] = v;
            }
        }
}

// ---------------------------------------------------------------------------
// Fused attention (NO softmax, per reference):
//   O = where(mask, QK^T/scale + rel_bias, -inf) @ V     -> bf16 context
// One wave per (b, h, 16-row i-tile). j loop in steps of 32:
//   scores via wmma (K=DH=64), epilogue, LDS round-trip to A-fragment
//   layout (ds_store_b16 / ds_load_b128), then S@V wmma into 4 DH tiles.
// ---------------------------------------------------------------------------
__global__ __launch_bounds__(32) void attn_wmma(const bf16* __restrict__ qb,
                                                const bf16* __restrict__ kb,
                                                const bf16* __restrict__ vb,
                                                const int* __restrict__ mask,
                                                const float* __restrict__ relb,
                                                const float* __restrict__ scalep,
                                                bf16* __restrict__ outO) {
    __shared__ bf16 sS[16 * 32];             // one 16x32 bf16 score tile
    const int lane = threadIdx.x & 31;
    int tile = blockIdx.x;                   // ((b*H + h) * ITILES + it)
    const int ITILES = SSD / 16;
    const int it = tile % ITILES; tile /= ITILES;
    const int hh = tile % HH;
    const int b  = tile / HH;
    const int i0 = it * 16;

    const int colq = lane & 15;
    const int grpA = (lane < 16) ? 0 : 16;
    const int grpB = (lane < 16) ? 0 : 16;
    const int grpC = (lane < 16) ? 0 : 8;
    const float inv_scale = 1.0f / scalep[0];

    // Q fragments for K = DH = 64 -> two 16x32 bf16 A-fragments (b128 loads)
    Frag aQ[2];
    {
        const bf16* qp = qb + (size_t)(b * SSD + i0 + colq) * DD + hh * DHD + grpA;
        #pragma unroll
        for (int t = 0; t < 2; ++t) {
            aQ[t].h[0] = *(const ushort8*)(qp + t * 32);
            aQ[t].h[1] = *(const ushort8*)(qp + t * 32 + 8);
        }
    }
    int mi8[8];
    #pragma unroll
    for (int r = 0; r < 8; ++r) mi8[r] = mask[b * SSD + i0 + r + grpC];

    v8f oacc[4];
    #pragma unroll
    for (int f = 0; f < 4; ++f) oacc[f] = zero8();

    for (int j0 = 0; j0 < SSD; j0 += 32) {
        // two 16x16 score sub-tiles -> LDS as bf16
        #pragma unroll
        for (int jj = 0; jj < 2; ++jj) {
            const int jcol = j0 + jj * 16 + colq;          // per-lane column j
            const bf16* kp = kb + (size_t)(b * SSD + jcol) * DD + hh * DHD + grpB;
            v8f c = zero8();
            #pragma unroll
            for (int t = 0; t < 2; ++t) {
                Frag bk;
                bk.h[0] = *(const ushort8*)(kp + t * 32);
                bk.h[1] = *(const ushort8*)(kp + t * 32 + 8);
                c = __builtin_amdgcn_wmma_f32_16x16x32_bf16(
                    false, aQ[t].v, false, bk.v, (short)0, c, false, false);
            }
            const int mj = mask[b * SSD + jcol];
            #pragma unroll
            for (int r = 0; r < 8; ++r) {
                int ii = i0 + r + grpC;
                float s = c[r] * inv_scale + relb[(jcol - ii + SSD - 1) * HH + hh];
                if (!(mi8[r] && mj)) s = -__builtin_inff();
                sS[(r + grpC) * 32 + jj * 16 + colq] = (bf16)s;
            }
        }
        asm volatile("" ::: "memory");   // keep LDS stores before reads (DS in-order)

        // score tile as 16x32 A-fragment (two ds_load_b128)
        Frag aS;
        {
            const bf16* p = &sS[colq * 32 + grpA];
            aS.h[0] = *(const ushort8*)(p);
            aS.h[1] = *(const ushort8*)(p + 8);
        }
        asm volatile("" ::: "memory");

        // O += S @ V over 4 column tiles of DH (int offsets -> imm-offset loads)
        const bf16* vstep = vb + (size_t)(b * SSD + j0 + grpB) * DD + hh * DHD + colq;
        #pragma unroll
        for (int f = 0; f < 4; ++f) {
            Frag bV;
            #pragma unroll
            for (int s = 0; s < 16; ++s) bV.e[s] = vstep[f * 16 + s * DD];
            oacc[f] = __builtin_amdgcn_wmma_f32_16x16x32_bf16(
                false, aS.v, false, bV.v, (short)0, oacc[f], false, false);
        }
    }

    #pragma unroll
    for (int f = 0; f < 4; ++f)
        #pragma unroll
        for (int r = 0; r < 8; ++r) {
            int ii = i0 + r + grpC;
            outO[(size_t)(b * SSD + ii) * DD + hh * DHD + f * 16 + colq] =
                (bf16)oacc[f][r];
        }
}

// ---------------------------------------------------------------------------
extern "C" void kernel_launch(void* const* d_in, const int* in_sizes, int n_in,
                              void* d_out, int out_size, void* d_ws, size_t ws_size,
                              hipStream_t stream) {
    const int*   x    = (const int*)d_in[0];
    const int*   mask = (const int*)d_in[1];
    const float* emb  = (const float*)d_in[2];
    const float* pos  = (const float*)d_in[3];
    const float* Wq   = (const float*)d_in[4];
    const float* bq   = (const float*)d_in[5];
    const float* Wk   = (const float*)d_in[6];
    const float* bk   = (const float*)d_in[7];
    const float* Wv   = (const float*)d_in[8];
    const float* bv   = (const float*)d_in[9];
    const float* Wo   = (const float*)d_in[10];
    const float* bo   = (const float*)d_in[11];
    const float* relb = (const float*)d_in[12];
    const float* scal = (const float*)d_in[13];
    const float* n1w  = (const float*)d_in[14];
    const float* n1b  = (const float*)d_in[15];
    const float* n2w  = (const float*)d_in[16];
    const float* n2b  = (const float*)d_in[17];
    const float* f1w  = (const float*)d_in[18];
    const float* f1b  = (const float*)d_in[19];
    const float* f2w  = (const float*)d_in[20];
    const float* f2b  = (const float*)d_in[21];
    const float* fnw  = (const float*)d_in[22];
    const float* fnb  = (const float*)d_in[23];
    const float* fcw  = (const float*)d_in[24];
    const float* fcb  = (const float*)d_in[25];

    char* ws = (char*)d_ws;
    float* h32   = (float*)ws;  ws += (size_t)MMD * DD * 4;   // 8 MB
    float* tmp2  = (float*)ws;  ws += (size_t)MMD * DD * 4;   // 8 MB
    bf16*  h16   = (bf16*)ws;   ws += (size_t)MMD * DD * 2;   // 4 MB
    bf16*  qbuf  = (bf16*)ws;   ws += (size_t)MMD * DD * 2;   // 4 MB
    bf16*  kbuf  = (bf16*)ws;   ws += (size_t)MMD * DD * 2;   // 4 MB
    bf16*  vbuf  = (bf16*)ws;   ws += (size_t)MMD * DD * 2;   // 4 MB
    bf16*  ctx16 = (bf16*)ws;   ws += (size_t)MMD * DD * 2;   // 4 MB
    bf16*  ffmid = (bf16*)ws;   ws += (size_t)MMD * FFD * 2;  // 16 MB
    bf16*  wbuf  = (bf16*)ws;   ws += (size_t)DD * VVD * 2;   // 32 MB (max K*N)

    embed_kernel<<<MMD, 256, 0, stream>>>(x, emb, pos, h32, h16);

    const int wavesD = (MMD / 32) * (DD / 64);    // 1024
    const int blksD  = (wavesD + 7) / 8;
    const int wavesF = (MMD / 32) * (FFD / 64);   // 4096
    const int blksF  = (wavesF + 7) / 8;
    const dim3 ctD(DD / 32, DD / 32);             // (K=512, N=512) weights
    const dim3 ctF1(FFD / 32, DD / 32);           // (K=512, N=2048)
    const dim3 ctF2(DD / 32, FFD / 32);           // (K=2048, N=512)

    for (int l = 0; l < LLD; ++l) {
        const float* bql = bq + (size_t)l * DD;
        const float* bkl = bk + (size_t)l * DD;
        const float* bvl = bv + (size_t)l * DD;
        const float* bol = bo + (size_t)l * DD;
        const float* rbl = relb + (size_t)l * (2 * SSD - 1) * HH;

        // QKV projections (bf16 A, transposed bf16 W, bf16 out)
        convt_kernel<<<ctD, 256, 0, stream>>>(Wq + (size_t)l * DD * DD, wbuf, DD, DD);
        gemm_wmma<false, true><<<blksD, 256, 0, stream>>>(h16, wbuf, bql, qbuf, MMD, DD, DD);
        convt_kernel<<<ctD, 256, 0, stream>>>(Wk + (size_t)l * DD * DD, wbuf, DD, DD);
        gemm_wmma<false, true><<<blksD, 256, 0, stream>>>(h16, wbuf, bkl, kbuf, MMD, DD, DD);
        convt_kernel<<<ctD, 256, 0, stream>>>(Wv + (size_t)l * DD * DD, wbuf, DD, DD);
        gemm_wmma<false, true><<<blksD, 256, 0, stream>>>(h16, wbuf, bvl, vbuf, MMD, DD, DD);

        // fused attention -> ctx16 (bf16 context)
        attn_wmma<<<BBD * HH * (SSD / 16), 32, 0, stream>>>(qbuf, kbuf, vbuf, mask, rbl,
                                                            scal + l, ctx16);

        // output projection -> tmp2 (f32), then h = LN(h + tmp2) (f32 + bf16)
        convt_kernel<<<ctD, 256, 0, stream>>>(Wo + (size_t)l * DD * DD, wbuf, DD, DD);
        gemm_wmma<false, false><<<blksD, 256, 0, stream>>>(ctx16, wbuf, bol, tmp2, MMD, DD, DD);
        ln_kernel<<<MMD, 256, 0, stream>>>(h32, tmp2, n1w + (size_t)l * DD,
                                           n1b + (size_t)l * DD, h32, h16);

        // FFN: gelu(h@f1 + b1) -> ffmid (bf16), @ f2 + b2 -> tmp2, h = LN(h + tmp2)
        convt_kernel<<<ctF1, 256, 0, stream>>>(f1w + (size_t)l * DD * FFD, wbuf, DD, FFD);
        gemm_wmma<true, true><<<blksF, 256, 0, stream>>>(h16, wbuf, f1b + (size_t)l * FFD,
                                                         ffmid, MMD, FFD, DD);
        convt_kernel<<<ctF2, 256, 0, stream>>>(f2w + (size_t)l * FFD * DD, wbuf, FFD, DD);
        gemm_wmma<false, false><<<blksD, 256, 0, stream>>>(ffmid, wbuf, f2b + (size_t)l * DD,
                                                           tmp2, MMD, DD, FFD);
        ln_kernel<<<MMD, 256, 0, stream>>>(h32, tmp2, n2w + (size_t)l * DD,
                                           n2b + (size_t)l * DD, h32, h16);
    }

    // final LN (bf16 only, reuse qbuf), then logits = LN(h) @ fcw + fcb -> d_out
    ln_kernel<<<MMD, 256, 0, stream>>>(h32, nullptr, fnw, fnb, nullptr, qbuf);
    convt_kernel<<<dim3(VVD / 32, DD / 32), 256, 0, stream>>>(fcw, wbuf, DD, VVD);
    const int wavesV = (MMD / 32) * (VVD / 64);   // 64000
    gemm_wmma<false, false><<<(wavesV + 7) / 8, 256, 0, stream>>>(
        qbuf, wbuf, fcb, (float*)d_out, MMD, VVD, DD);
}
